// MHARawEncoder_74148315398510
// MI455X (gfx1250) — matrix operand order repaired and checked by reading, not compiled, not run on hardware
//
#include <hip/hip_runtime.h>
#include <hip/hip_bf16.h>
#include <stdint.h>

#define EMBED 512
#define HEADS 8
#define KNN   5
#define BQ    8192
#define NMEM  50000
#define NSEG  8
#define SEGLEN 6250      // NMEM / NSEG
#define SC_WAVES 8       // waves per block in the scores kernel
#define BPITCH 520       // LDS row pitch (elements): 260 dwords -> conflict-free

typedef __attribute__((ext_vector_type(16))) __bf16 v16bf;
typedef __attribute__((ext_vector_type(8)))  float  v8f;
typedef __attribute__((ext_vector_type(2)))  float  v2f;

union FragBF { v16bf v; uint4 q[2]; };

__device__ __forceinline__ unsigned short f2bf_rne(float x) {
    unsigned int u = __float_as_uint(x);
    unsigned int r = u + 0x7FFFu + ((u >> 16) & 1u);   // round-to-nearest-even
    return (unsigned short)(r >> 16);
}

// ---------------------------------------------------------------- prep kernels
__global__ void k_conv_bf16(const float* __restrict__ src,
                            unsigned short* __restrict__ dst, int n) {
    int i = blockIdx.x * blockDim.x + threadIdx.x;
    if (i < n) dst[i] = f2bf_rne(src[i]);
}

__global__ void k_rowsq(const float* __restrict__ mk, float* __restrict__ sq, int rows) {
    int r = blockIdx.x * blockDim.x + threadIdx.x;
    if (r >= rows) return;
    const float* p = mk + (size_t)r * EMBED;
    float s = 0.f;
    for (int c = 0; c < EMBED; ++c) { float v = p[c]; s += v * v; }
    sq[r] = s;
}

// Wc = Win @ Wx  (both 512x512 row-major);  bc = Win @ bx + b_in_seg
__global__ void k_combine(const float* __restrict__ Win, const float* __restrict__ Wx,
                          const float* __restrict__ bx, const float* __restrict__ b_in_seg,
                          float* __restrict__ Wc, float* __restrict__ bc) {
    int t = blockIdx.x * blockDim.x + threadIdx.x;   // 512*512 threads
    int n = t >> 9, k = t & 511;
    const float* wr = Win + (size_t)n * EMBED;
    float s = 0.f;
    for (int j = 0; j < EMBED; ++j) s += wr[j] * Wx[(size_t)j * EMBED + k];
    Wc[(size_t)n * EMBED + k] = s;
    if (k == 0) {
        float b = b_in_seg[n];
        for (int j = 0; j < EMBED; ++j) b += wr[j] * bx[j];
        bc[n] = b;
    }
}

// ---------------------------------------------- bf16 WMMA scores + fused top-5
// 8 waves / block, 128 query rows / block; one shared 16-column key tile per
// iteration, double-buffered in LDS via async global->LDS loads (ASYNCcnt).
// LDS->register B fragments pipelined 2 chunks ahead; sched_barrier(0) pins
// the issue order so the waitcnt pass emits s_wait_dscnt<=4 per WMMA.
__global__ void __launch_bounds__(256)
k_scores_topk(const unsigned short* __restrict__ Qbf,
              const unsigned short* __restrict__ Kbf,
              const float* __restrict__ sq,
              float* __restrict__ pval, int* __restrict__ pidx) {
    const int tid   = threadIdx.x;
    const int lane  = tid & 31;
    const int wave  = tid >> 5;
    const int m0    = (blockIdx.x * SC_WAVES + wave) * 16;
    const int seg   = blockIdx.y;
    const int jstart = seg * SEGLEN;
    const int jend   = (jstart + SEGLEN < NMEM) ? (jstart + SEGLEN) : NMEM;
    const int row16 = lane & 15;
    const int hi    = lane >> 4;

    __shared__ __align__(16) unsigned short Bt[2][16][BPITCH];  // 2 x 16.25KB
    __shared__ float sc[SC_WAVES][16][17];

    // A fragments (16 rows x 512 K, bf16) resident in registers.
    // ISA A layout: lanes 0-15 hold K {c*32+0..7, c*32+16..23}, lanes 16-31 +8.
    v16bf aq[16];
    {
        const unsigned short* qrow = Qbf + (size_t)(m0 + row16) * EMBED;
        #pragma unroll
        for (int kc = 0; kc < 16; ++kc) {
            FragBF f;
            int base = kc * 32 + hi * 8;
            f.q[0] = *(const uint4*)(qrow + base);
            f.q[1] = *(const uint4*)(qrow + base + 16);
            aq[kc] = f.v;
        }
    }

    float tv[KNN]; int ti[KNN];
    #pragma unroll
    for (int i = 0; i < KNN; ++i) { tv[i] = -3.0e38f; ti[i] = 0; }

    const int ntiles = (jend - jstart + 15) >> 4;

    // ---- stage tile 0 into Bt[0]: 16 rows x 1KB, async global->LDS ----
    #pragma unroll
    for (int r = 0; r < 4; ++r) {
        int i   = tid + r * 256;        // 0..1023 b128 transfers
        int row = i >> 6;               // 64 x 16B per 1KB row
        int col = i & 63;
        int j   = jstart + row;
        const unsigned short* g = Kbf + (size_t)(j < NMEM ? j : 0) * EMBED + col * 8;
        unsigned int lo = (unsigned int)(uintptr_t)(void*)&Bt[0][row][col * 8];
        asm volatile("global_load_async_to_lds_b128 %0, %1, off"
                     :: "v"(lo), "v"(g) : "memory");
    }
    asm volatile("s_wait_asynccnt 0x0" ::: "memory");
    __syncthreads();

    for (int t = 0; t < ntiles; ++t) {
        const int cur   = t & 1;
        const int jbase = jstart + t * 16;

        // ---- async-stage tile t+1 into the other buffer ----
        if (t + 1 < ntiles) {
            const int jb2 = jbase + 16;
            #pragma unroll
            for (int r = 0; r < 4; ++r) {
                int i   = tid + r * 256;
                int row = i >> 6;
                int col = i & 63;
                int j   = jb2 + row;
                const unsigned short* g =
                    Kbf + (size_t)(j < NMEM ? j : 0) * EMBED + col * 8;
                unsigned int lo =
                    (unsigned int)(uintptr_t)(void*)&Bt[cur ^ 1][row][col * 8];
                asm volatile("global_load_async_to_lds_b128 %0, %1, off"
                             :: "v"(lo), "v"(g) : "memory");
            }
        }

        // ---- compute 16x16 score tile from LDS (conflict-free pitch) ----
        // B layout: lane = column (row16), K {hi*8+0..7, 16+hi*8+0..7} per chunk.
        const unsigned short* brow = &Bt[cur][row16][hi * 8];
        v8f c = {};
        FragBF fb[3];
        fb[0].q[0] = *(const uint4*)(brow);
        fb[0].q[1] = *(const uint4*)(brow + 16);
        fb[1].q[0] = *(const uint4*)(brow + 32);
        fb[1].q[1] = *(const uint4*)(brow + 48);
        __builtin_amdgcn_sched_barrier(0);
        #pragma unroll
        for (int kc = 0; kc < 16; ++kc) {
            if (kc + 2 < 16) {                       // prefetch chunk kc+2
                FragBF* f = &fb[(kc + 2) % 3];
                f->q[0] = *(const uint4*)(brow + (kc + 2) * 32);
                f->q[1] = *(const uint4*)(brow + (kc + 2) * 32 + 16);
            }
            c = __builtin_amdgcn_wmma_f32_16x16x32_bf16(
                    false, aq[kc], false, fb[kc % 3].v, (short)0, c, false, false);
            __builtin_amdgcn_sched_barrier(0);       // pin: loads stay 2 ahead
        }

        // ---- per-wave epilogue: C tile -> LDS -> per-row top-5 ----
        #pragma unroll
        for (int r = 0; r < 8; ++r)           // C: lane=col, vgpr r = row r+8*hi
            sc[wave][hi * 8 + r][row16] = c[r];
        asm volatile("s_wait_dscnt 0x0" ::: "memory");  // wave-internal RAW

        if (lane < 16) {                      // lane owns query row m0+lane
            const float* srow = sc[wave][lane];
            for (int c2 = 0; c2 < 16; ++c2) {
                int j = jbase + c2;
                if (j >= jend) break;
                float s = 2.0f * srow[c2] - sq[j];
                if (s > tv[KNN - 1]) {
                    int p = KNN - 1;
                    while (p > 0 && s > tv[p - 1]) {
                        tv[p] = tv[p - 1]; ti[p] = ti[p - 1]; --p;
                    }
                    tv[p] = s; ti[p] = j;
                }
            }
        }

        // own async stage done; all waves finished reading Bt[cur]
        asm volatile("s_wait_asynccnt 0x0" ::: "memory");
        __syncthreads();
    }

    if (lane < 16) {
        int m = m0 + lane;
        size_t off = ((size_t)m * NSEG + seg) * KNN;
        #pragma unroll
        for (int i = 0; i < KNN; ++i) { pval[off + i] = tv[i]; pidx[off + i] = ti[i]; }
    }
}

__global__ void k_merge(const float* __restrict__ pval, const int* __restrict__ pidx,
                        int* __restrict__ idx) {
    int m = blockIdx.x * blockDim.x + threadIdx.x;
    if (m >= BQ) return;
    float tv[KNN]; int ti[KNN];
    for (int i = 0; i < KNN; ++i) { tv[i] = -3.0e38f; ti[i] = 0; }
    for (int s = 0; s < NSEG; ++s) {
        size_t off = ((size_t)m * NSEG + s) * KNN;
        for (int i = 0; i < KNN; ++i) {
            float v = pval[off + i]; int j = pidx[off + i];
            if (v > tv[KNN - 1]) {
                int p = KNN - 1;
                while (p > 0 && v > tv[p - 1]) { tv[p] = tv[p - 1]; ti[p] = ti[p - 1]; --p; }
                tv[p] = v; ti[p] = j;
            }
        }
    }
    for (int i = 0; i < KNN; ++i) idx[m * KNN + i] = ti[i];
}

// ------------------------------- exact fp32 WMMA GEMM: C = A@W^T + b
// W row-major (N x K). Optional row gather: A row m taken from gbase[gidx[m]].
// Loads pinned 2 steps ahead of their WMMA via sched_barrier(0).
__global__ void __launch_bounds__(32)
k_gemm_wmma_f32(const float* __restrict__ A, const float* __restrict__ W,
                const float* __restrict__ bias, float* __restrict__ C,
                int N, int K,
                const int* __restrict__ gidx, const float* __restrict__ gbase) {
    const int lane  = threadIdx.x;
    const int row16 = lane & 15;
    const int hi    = lane >> 4;
    const int n0    = blockIdx.x * 16;
    const int m0    = blockIdx.y * 16;
    const int m     = m0 + row16;

    const float* arow = gidx ? (gbase + (size_t)gidx[m] * K)
                             : (A + (size_t)m * K);
    const float* wrow = W + (size_t)(n0 + row16) * K;
    const int ko = hi * 2;   // f32 16x16x4: lanes 0-15 K{0,1}, lanes 16-31 K{2,3}

    v8f c = {};
    v2f a[3], b[3];
    a[0] = *(const v2f*)(arow + ko);
    b[0] = *(const v2f*)(wrow + ko);
    a[1] = *(const v2f*)(arow + 4 + ko);
    b[1] = *(const v2f*)(wrow + 4 + ko);
    __builtin_amdgcn_sched_barrier(0);
    #pragma unroll 8
    for (int k = 0; k < K; k += 4) {
        int step = k >> 2;
        int kn = (k + 8 < K) ? (k + 8) : 0;      // clamp: no OOB, no branch
        a[(step + 2) % 3] = *(const v2f*)(arow + kn + ko);
        b[(step + 2) % 3] = *(const v2f*)(wrow + kn + ko);
        c = __builtin_amdgcn_wmma_f32_16x16x4_f32(
                false, a[step % 3], false, b[step % 3], (short)0, c, false, false);
        __builtin_amdgcn_sched_barrier(0);       // pin: loads stay 2 ahead
    }

    const int col = n0 + row16;
    const float bb = bias ? bias[col] : 0.0f;
    const int rb = hi * 8;
    #pragma unroll
    for (int r = 0; r < 8; ++r)
        C[(size_t)(m0 + rb + r) * N + col] = c[r] + bb;
}

// ---------------------------------------------- tiny 5-key softmax attention
__global__ void k_attn(const float* __restrict__ qe, const float* __restrict__ ke,
                       const float* __restrict__ ve, float* __restrict__ out) {
    int t = blockIdx.x * blockDim.x + threadIdx.x;
    if (t >= BQ * HEADS) return;
    int q = t >> 3, h = t & 7;
    const float* qh = qe + (size_t)q * EMBED + h * 64;
    float s[KNN];
    float mx = -3.0e38f;
    for (int j = 0; j < KNN; ++j) {
        const float* kh = ke + ((size_t)q * KNN + j) * EMBED + h * 64;
        float d = 0.f;
        for (int c = 0; c < 64; ++c) d += qh[c] * kh[c];
        s[j] = d * 0.125f;                       // 1/sqrt(64)
        mx = fmaxf(mx, s[j]);
    }
    float sum = 0.f;
    for (int j = 0; j < KNN; ++j) { s[j] = expf(s[j] - mx); sum += s[j]; }
    float inv = 1.0f / sum;
    float o[64];
    for (int c = 0; c < 64; ++c) o[c] = 0.f;
    for (int j = 0; j < KNN; ++j) {
        const float* vh = ve + ((size_t)q * KNN + j) * EMBED + h * 64;
        float a = s[j] * inv;
        for (int c = 0; c < 64; ++c) o[c] += a * vh[c];
    }
    float* op = out + (size_t)q * EMBED + h * 64;
    for (int c = 0; c < 64; ++c) op[c] = o[c];
}

// ------------------------------------------------------------------- launcher
extern "C" void kernel_launch(void* const* d_in, const int* in_sizes, int n_in,
                              void* d_out, int out_size, void* d_ws, size_t ws_size,
                              hipStream_t stream) {
    const float* Q    = (const float*)d_in[0];
    const float* MK   = (const float*)d_in[1];
    const float* MV   = (const float*)d_in[2];
    const float* Wq   = (const float*)d_in[3];
    const float* bq   = (const float*)d_in[4];
    const float* Wk   = (const float*)d_in[5];
    const float* bk   = (const float*)d_in[6];
    const float* Wv   = (const float*)d_in[7];
    const float* bv   = (const float*)d_in[8];
    const float* Winq = (const float*)d_in[9];
    const float* Wink = (const float*)d_in[10];
    const float* Winv = (const float*)d_in[11];
    const float* b_in = (const float*)d_in[12];
    const float* Wout = (const float*)d_in[13];
    const float* bout = (const float*)d_in[14];
    float* out = (float*)d_out;

    char* ws = (char*)d_ws;
    size_t off = 0;
    auto carve = [&](size_t bytes) -> void* {
        off = (off + 255) & ~(size_t)255;
        void* p = ws + off;
        off += bytes;
        return p;
    };
    unsigned short* Qbf = (unsigned short*)carve((size_t)BQ * EMBED * 2);
    unsigned short* Kbf = (unsigned short*)carve((size_t)NMEM * EMBED * 2);
    float* sq   = (float*)carve((size_t)NMEM * 4);
    float* Wcq  = (float*)carve((size_t)EMBED * EMBED * 4);
    float* Wck  = (float*)carve((size_t)EMBED * EMBED * 4);
    float* Wcv  = (float*)carve((size_t)EMBED * EMBED * 4);
    float* bcq  = (float*)carve((size_t)EMBED * 4);
    float* bck  = (float*)carve((size_t)EMBED * 4);
    float* bcv  = (float*)carve((size_t)EMBED * 4);
    float* pval = (float*)carve((size_t)BQ * NSEG * KNN * 4);
    int*   pidx = (int*)  carve((size_t)BQ * NSEG * KNN * 4);
    int*   idx  = (int*)  carve((size_t)BQ * KNN * 4);
    float* qe   = (float*)carve((size_t)BQ * EMBED * 4);
    float* kep  = (float*)carve((size_t)BQ * KNN * EMBED * 4);
    float* vep  = (float*)carve((size_t)BQ * KNN * EMBED * 4);
    float* ao   = (float*)carve((size_t)BQ * EMBED * 4);

    // Stage 0: conversions + row norms + folded weights
    int nQ = BQ * EMBED;
    k_conv_bf16<<<(nQ + 255) / 256, 256, 0, stream>>>(Q, Qbf, nQ);
    int nK = NMEM * EMBED;
    k_conv_bf16<<<(nK + 255) / 256, 256, 0, stream>>>(MK, Kbf, nK);
    k_rowsq<<<(NMEM + 127) / 128, 128, 0, stream>>>(MK, sq, NMEM);
    k_combine<<<(EMBED * EMBED) / 256, 256, 0, stream>>>(Winq, Wq, bq, b_in,             Wcq, bcq);
    k_combine<<<(EMBED * EMBED) / 256, 256, 0, stream>>>(Wink, Wk, bk, b_in + EMBED,     Wck, bck);
    k_combine<<<(EMBED * EMBED) / 256, 256, 0, stream>>>(Winv, Wv, bv, b_in + 2 * EMBED, Wcv, bcv);

    // Stage 1: bf16 WMMA similarity + fused per-segment top-5, then merge
    dim3 gs(BQ / (16 * SC_WAVES), NSEG);
    k_scores_topk<<<gs, 256, 0, stream>>>(Qbf, Kbf, sq, pval, pidx);
    k_merge<<<(BQ + 255) / 256, 256, 0, stream>>>(pval, pidx, idx);

    // Stage 2: exact fp32 WMMA projections (K/V gathered on the fly)
    dim3 g1(EMBED / 16, BQ / 16);
    dim3 g2(EMBED / 16, (BQ * KNN) / 16);
    k_gemm_wmma_f32<<<g1, 32, 0, stream>>>(Q,  Wcq, bcq, qe,  EMBED, EMBED, nullptr, nullptr);
    k_gemm_wmma_f32<<<g2, 32, 0, stream>>>(nullptr, Wck, bck, kep, EMBED, EMBED, idx, MK);
    k_gemm_wmma_f32<<<g2, 32, 0, stream>>>(nullptr, Wcv, bcv, vep, EMBED, EMBED, idx, MV);

    // Stage 3: softmax attention over 5 keys, then output projection
    k_attn<<<(BQ * HEADS + 255) / 256, 256, 0, stream>>>(qe, kep, vep, ao);
    k_gemm_wmma_f32<<<g1, 32, 0, stream>>>(ao, Wout, bout, out, EMBED, EMBED, nullptr, nullptr);
}
